// RNNWavefunction_8108898254939
// MI455X (gfx1250) — matrix operand order repaired and checked by reading, not compiled, not run on hardware
//
#include <hip/hip_runtime.h>
#include <hip/hip_bf16.h>

// ---------------------------------------------------------------------------
// RNN wavefunction (2-layer GRU, H=256) on gfx1250 using bf16 WMMA.
//   B=4096, SORB=64 steps, H=256, 3H=768 gate dims, NELE=32 -> alpha=16.
// One workgroup (256 threads = 8 wave32) owns BM=64 batch rows and runs the
// whole 64-step recurrence with hidden state ping-ponged in LDS.
// Weights are prepacked to bf16 in d_ws:
//   w0 : Whh0            (768 x 256)  row-major (N x K)
//   w1 : [Wih1 | Whh1]   (768 x 512)  row-major (N x K), fused K=512 GEMM
// ---------------------------------------------------------------------------

typedef __attribute__((ext_vector_type(16))) __bf16 v16bf;
typedef __attribute__((ext_vector_type(8)))  __bf16 v8bf;
typedef __attribute__((ext_vector_type(8)))  float  v8f;

#define HID   256
#define G3    768
#define BM    64
#define SORB_ 64
#define NSTEP 64

__device__ __forceinline__ float sigmoidf_(float x) {
  return 1.0f / (1.0f + __expf(-x));
}

// A-fragment (16x32 bf16, M x K), row-major source with stride HID.
// Lane L (<16): row=L, K = 0..15; lane L>=16: row=L-16,
// K = 16..31.  Two contiguous 16B LDS loads per lane.
__device__ __forceinline__ v16bf load_a_frag(const __bf16* hbase, int lane, int kbase) {
  int row  = lane & 15;
  int koff = (lane >> 4) * 16;
  const __bf16* p = hbase + row * HID + kbase + koff;
  v8bf lo = *(const v8bf*)(p);
  v8bf hi = *(const v8bf*)(p + 8);
  return __builtin_shufflevector(lo, hi, 0,1,2,3,4,5,6,7,8,9,10,11,12,13,14,15);
}

// B-fragment (32x16 bf16, K x N) from W stored row-major as (N x K).
// Lane L (<16): col=L, K = kbase+0..15; lane L>=16: col=L-16, K = kbase+16..31.
// 32 contiguous bytes per lane (two 16B global loads).
__device__ __forceinline__ v16bf load_b_frag(const __bf16* wbase, int lane,
                                             int kbase, int stride) {
  int col  = lane & 15;
  int krow = (lane >> 4) * 16;
  const __bf16* p = wbase + col * stride + kbase + krow;
  v8bf lo = *(const v8bf*)(p);
  v8bf hi = *(const v8bf*)(p + 8);
  return __builtin_shufflevector(lo, hi, 0,1,2,3,4,5,6,7,8,9,10,11,12,13,14,15);
}

__device__ __forceinline__ v8f wmma_bf16(v16bf a, v16bf b, v8f c) {
  return __builtin_amdgcn_wmma_f32_16x16x32_bf16(false, a, false, b,
                                                 (short)0, c, false, false);
}

// ---------------------------------------------------------------------------
// Weight prepack: f32 -> bf16, fused [Wih1 | Whh1] for layer 1.
// ---------------------------------------------------------------------------
__global__ void rnnwf_prepack(const float* __restrict__ Whh0,
                              const float* __restrict__ Wih1,
                              const float* __restrict__ Whh1,
                              __bf16* __restrict__ w0,
                              __bf16* __restrict__ w1) {
  int i = blockIdx.x * 256 + threadIdx.x;
  if (i < G3 * HID) w0[i] = (__bf16)Whh0[i];
  if (i < G3 * 2 * HID) {
    int n = i >> 9;        // / 512
    int k = i & 511;
    float v = (k < HID) ? Wih1[n * HID + k] : Whh1[n * HID + (k - HID)];
    w1[i] = (__bf16)v;
  }
}

// ---------------------------------------------------------------------------
// Main recurrent kernel.
// ---------------------------------------------------------------------------
__global__ __launch_bounds__(256)
void rnnwf_kernel(const int*   __restrict__ xg,    // (B, 64) in {-1,+1}
                  const float* __restrict__ Wih0,  // (768, 2)
                  const float* __restrict__ Wl,    // (2, 256)
                  const float* __restrict__ bl,    // (2,)
                  const __bf16* __restrict__ w0,   // (768, 256) bf16
                  const __bf16* __restrict__ w1,   // (768, 512) bf16
                  float* __restrict__ out) {       // (B, 2) re/im interleaved
  __shared__ __align__(16) __bf16 sH0[2][BM * HID];  // 64 KB ping-pong
  __shared__ __align__(16) __bf16 sH1[2][BM * HID];  // 64 KB ping-pong
  __shared__ float sWih0[G3 * 2];                    // 6 KB
  __shared__ float sWl[2 * HID];                     // 2 KB
  __shared__ float sBl[2];
  __shared__ int   sPrev[BM];                        // previous bit (2 = none)

  const int tid  = threadIdx.x;
  const int lane = tid & 31;
  const int wave = tid >> 5;
  const int grow = blockIdx.x * BM + tid;            // valid when tid < BM

  for (int i = tid; i < G3 * 2; i += 256) sWih0[i] = Wih0[i];
  for (int i = tid; i < 2 * HID; i += 256) sWl[i] = Wl[i];
  if (tid < 2)  sBl[tid] = bl[tid];
  if (tid < BM) sPrev[tid] = 2;
  for (int i = tid; i < BM * HID; i += 256) {
    sH0[0][i] = (__bf16)0.0f;
    sH1[0][i] = (__bf16)0.0f;
  }
  __syncthreads();

  float amp = 1.0f, phase = 0.0f, nup = 0.0f, ndn = 0.0f;

  #pragma unroll 1
  for (int s = 0; s < NSTEP; ++s) {
    const __bf16* h0c = sH0[s & 1];
    __bf16*       h0n = sH0[(s & 1) ^ 1];
    const __bf16* h1c = sH1[s & 1];
    __bf16*       h1n = sH1[(s & 1) ^ 1];

    // ---------------- Layer 0: gates = h0 @ Whh0^T (+ one-hot Wih0 pick) ---
    #pragma unroll 1
    for (int t = 0; t < 8; ++t) {
      const int nt = wave * 2 + (t >> 2);
      const int mt = t & 3;
      v8f ar = {}, az = {}, ah = {};
      const __bf16* abase = h0c + mt * 16 * HID;
      const __bf16* b0r = w0 + (0 * HID + nt * 16) * HID;
      const __bf16* b0z = w0 + (1 * HID + nt * 16) * HID;
      const __bf16* b0n = w0 + (2 * HID + nt * 16) * HID;
      #pragma unroll
      for (int kt = 0; kt < 8; ++kt) {
        v16bf a = load_a_frag(abase, lane, kt * 32);
        ar = wmma_bf16(a, load_b_frag(b0r, lane, kt * 32, HID), ar);
        az = wmma_bf16(a, load_b_frag(b0z, lane, kt * 32, HID), az);
        ah = wmma_bf16(a, load_b_frag(b0n, lane, kt * 32, HID), ah);
      }
      const int col = nt * 16 + (lane & 15);
      const float wr0 = sWih0[(col)*2 + 0],       wr1 = sWih0[(col)*2 + 1];
      const float wz0 = sWih0[(col + 256)*2 + 0], wz1 = sWih0[(col + 256)*2 + 1];
      const float wn0 = sWih0[(col + 512)*2 + 0], wn1 = sWih0[(col + 512)*2 + 1];
      const int rbase = mt * 16 + (lane >> 4) * 8;
      #pragma unroll
      for (int i = 0; i < 8; ++i) {
        const int row = rbase + i;
        const int pb  = sPrev[row];
        const float ir = (pb == 0) ? wr0 : ((pb == 1) ? wr1 : 0.0f);
        const float iz = (pb == 0) ? wz0 : ((pb == 1) ? wz1 : 0.0f);
        const float in = (pb == 0) ? wn0 : ((pb == 1) ? wn1 : 0.0f);
        const float r  = sigmoidf_(ir + ar[i]);
        const float z  = sigmoidf_(iz + az[i]);
        const float n  = tanhf(in + r * ah[i]);
        const float hp = (float)h0c[row * HID + col];
        h0n[row * HID + col] = (__bf16)((1.0f - z) * n + z * hp);
      }
    }
    __syncthreads();

    // ---------------- Layer 1: fused K=512, gi/gh n-gate kept separate -----
    #pragma unroll 1
    for (int t = 0; t < 8; ++t) {
      const int nt = wave * 2 + (t >> 2);
      const int mt = t & 3;
      v8f ar = {}, az = {}, ai = {}, ah = {};
      const __bf16* a0  = h0n + mt * 16 * HID;
      const __bf16* a1  = h1c + mt * 16 * HID;
      const __bf16* b1r = w1 + (0 * HID + nt * 16) * 512;
      const __bf16* b1z = w1 + (1 * HID + nt * 16) * 512;
      const __bf16* b1n = w1 + (2 * HID + nt * 16) * 512;
      #pragma unroll
      for (int kt = 0; kt < 8; ++kt) {           // K 0..255: A = h0n (Wih1 half)
        v16bf a = load_a_frag(a0, lane, kt * 32);
        ar = wmma_bf16(a, load_b_frag(b1r, lane, kt * 32, 512), ar);
        az = wmma_bf16(a, load_b_frag(b1z, lane, kt * 32, 512), az);
        ai = wmma_bf16(a, load_b_frag(b1n, lane, kt * 32, 512), ai);
      }
      #pragma unroll
      for (int kt = 0; kt < 8; ++kt) {           // K 256..511: A = h1 (Whh1 half)
        v16bf a = load_a_frag(a1, lane, kt * 32);
        ar = wmma_bf16(a, load_b_frag(b1r, lane, 256 + kt * 32, 512), ar);
        az = wmma_bf16(a, load_b_frag(b1z, lane, 256 + kt * 32, 512), az);
        ah = wmma_bf16(a, load_b_frag(b1n, lane, 256 + kt * 32, 512), ah);
      }
      const int col   = nt * 16 + (lane & 15);
      const int rbase = mt * 16 + (lane >> 4) * 8;
      #pragma unroll
      for (int i = 0; i < 8; ++i) {
        const int row = rbase + i;
        const float r  = sigmoidf_(ar[i]);
        const float z  = sigmoidf_(az[i]);
        const float n  = tanhf(ai[i] + r * ah[i]);
        const float hp = (float)h1c[row * HID + col];
        h1n[row * HID + col] = (__bf16)((1.0f - z) * n + z * hp);
      }
    }
    __syncthreads();

    // ---------------- Head: logits, symmetry masks, amp/phase -------------
    if (tid < BM) {
      float l0 = sBl[0], l1 = sBl[1];
      const __bf16* hr = h1n + tid * HID;
      #pragma unroll 4
      for (int k = 0; k < HID; ++k) {
        const float h = (float)hr[k];
        l0 += h * sWl[k];
        l1 += h * sWl[HID + k];
      }
      const int   bit  = (xg[grow * SORB_ + s] + 1) >> 1;
      const bool  even = (s & 1) == 0;
      const float cnt  = even ? nup : ndn;
      const float m_occ = (16.0f > cnt) ? 1.0f : 0.0f;
      const float lower = -16.0f + (float)(s >> 1);
      const float m_un  = (lower < cnt) ? 1.0f : 0.0f;
      const float mx = fmaxf(l0, l1);
      const float e0 = __expf(l0 - mx), e1 = __expf(l1 - mx);
      const float inv = 1.0f / (e0 + e1);
      const float a0v = sqrtf(e0 * inv) * m_un;
      const float a1v = sqrtf(e1 * inv) * m_occ;
      const float nrm = fmaxf(sqrtf(a0v * a0v + a1v * a1v), 1e-12f);
      amp *= (bit ? a1v : a0v) / nrm;
      const float lg = bit ? l1 : l0;
      phase += 3.14159265358979323f * lg / (1.0f + fabsf(lg));
      if (even) nup += (float)bit; else ndn += (float)bit;
      sPrev[tid] = bit;
    }
    __syncthreads();
  }

  if (tid < BM) {
    out[grow * 2 + 0] = amp * __cosf(phase);
    out[grow * 2 + 1] = amp * __sinf(phase);
  }
}

// ---------------------------------------------------------------------------
extern "C" void kernel_launch(void* const* d_in, const int* in_sizes, int n_in,
                              void* d_out, int out_size, void* d_ws, size_t ws_size,
                              hipStream_t stream) {
  const int*   x    = (const int*)d_in[0];     // (B, 64)
  const float* Wih0 = (const float*)d_in[1];   // (768, 2)
  const float* Whh0 = (const float*)d_in[2];   // (768, 256)
  const float* Wih1 = (const float*)d_in[3];   // (768, 256)
  const float* Whh1 = (const float*)d_in[4];   // (768, 256)
  const float* Wl   = (const float*)d_in[5];   // (2, 256)
  const float* bl   = (const float*)d_in[6];   // (2,)

  const int B = in_sizes[0] / SORB_;           // 4096

  __bf16* w0 = (__bf16*)d_ws;                  // 768*256 bf16
  __bf16* w1 = w0 + G3 * HID;                  // 768*512 bf16

  rnnwf_prepack<<<(G3 * 2 * HID + 255) / 256, 256, 0, stream>>>(Whh0, Wih1, Whh1, w0, w1);
  rnnwf_kernel<<<B / BM, 256, 0, stream>>>(x, Wih0, Wl, bl, w0, w1, (float*)d_out);
}
